// ShiftedPatchTokenization_37830071943539
// MI455X (gfx1250) — matrix operand order repaired and checked by reading, not compiled, not run on hardware
//
#include <hip/hip_runtime.h>
#include <hip/hip_bf16.h>

// ---------------------------------------------------------------------------
// ShiftedPatchTokenization for MI455X (gfx1250, wave32, WMMA)
//   K1: transpose+convert weights fp32[3840,768] -> f16[768,3840]
//   K2: fused shift-gather + patchify + LayerNorm -> A f16[12544,3840]
//   K3: WMMA GEMM  out[12544,768] = A @ W + bias   (f16 in, f32 accum)
// ---------------------------------------------------------------------------

typedef __attribute__((ext_vector_type(16))) _Float16 v16h;
typedef __attribute__((ext_vector_type(8)))  _Float16 v8h;
typedef __attribute__((ext_vector_type(8)))  float    v8f;

#define IMG_H   224
#define IMG_W   224
#define GRID14  14          // 224 / 16
#define DIN     3840        // 15 * 16 * 16
#define DMODEL  768
#define NTOK    196
#define BATCH   64
#define M_TOTAL (BATCH * NTOK)   // 12544

// ---------------------------------------------------------------------------
// K1: W[k,n] (fp32, row-major [3840,768]) -> Wt[n,k] (f16, row-major [768,3840])
//     coalesced f16 writes; strided reads are tiny (12 MB, L2-friendly)
// ---------------------------------------------------------------------------
__global__ __launch_bounds__(256) void k_wt_convert(const float* __restrict__ w,
                                                    _Float16* __restrict__ wt) {
  int idx = blockIdx.x * 256 + threadIdx.x;       // over 768*3840
  int n = idx / DIN;
  int k = idx - n * DIN;
  wt[idx] = (_Float16)w[(size_t)k * DMODEL + n];
}

// ---------------------------------------------------------------------------
// K2: one block per token (m = b*196 + hh*14 + ww), 256 threads.
// Thread tid -> (p1 = tid>>4, p2 = tid&15) -> pixel (y = p1*14+hh, x = p2*14+ww).
// Each thread produces all 15 channel values for its pixel:
//   identity (3), left-shift L (3, duplicated to 6), right-shift R (3, dup to 6)
//   R[y,x] = 13/14 * x[y, x+8] + 1/14 * x[y+1, x+8]   (zero pad)
//   L[y,x] =  1/14 * x[y-1,x-8] + 13/14 * x[y,  x-8]  (zero pad)
// Then block-reduce mean/var over 3840 and emit normalized f16 row of A.
// ---------------------------------------------------------------------------
__global__ __launch_bounds__(256) void k_patch_ln(const float* __restrict__ x,
                                                  const float* __restrict__ gamma,
                                                  const float* __restrict__ beta,
                                                  _Float16* __restrict__ A) {
  const int m  = blockIdx.x;
  const int b  = m / NTOK;
  const int t  = m - b * NTOK;
  const int hh = t / GRID14;
  const int ww = t - hh * GRID14;

  const int tid = threadIdx.x;
  const int p1  = tid >> 4;
  const int p2  = tid & 15;
  const int y   = p1 * GRID14 + hh;     // 0..223
  const int xx  = p2 * GRID14 + ww;     // 0..223

  const float WA = 13.0f / 14.0f;
  const float WB = 1.0f  / 14.0f;
  const float* xb = x + (size_t)b * 3 * IMG_H * IMG_W;

  float v[15];
  const int xm = xx - 8, xp = xx + 8;
  #pragma unroll
  for (int cc = 0; cc < 3; ++cc) {
    const float* pl = xb + (size_t)cc * IMG_H * IMG_W;
    float idv = pl[y * IMG_W + xx];
    float lv = 0.0f;
    if (xm >= 0) {
      lv = WA * pl[y * IMG_W + xm];
      if (y > 0) lv += WB * pl[(y - 1) * IMG_W + xm];
    }
    float rv = 0.0f;
    if (xp < IMG_W) {
      rv = WA * pl[y * IMG_W + xp];
      if (y < IMG_H - 1) rv += WB * pl[(y + 1) * IMG_W + xp];
    }
    v[cc]      = idv;   // channels 0..2 : identity
    v[3 + cc]  = lv;    // channels 3..5 : left
    v[6 + cc]  = rv;    // channels 6..8 : right
    v[9 + cc]  = lv;    // channels 9..11: left (dup)
    v[12 + cc] = rv;    // channels12..14: right (dup)
  }

  float sum = 0.0f, sq = 0.0f;
  #pragma unroll
  for (int q = 0; q < 15; ++q) { sum += v[q]; sq += v[q] * v[q]; }

  // wave32 butterfly reduce, then cross-wave via LDS
  #pragma unroll
  for (int off = 16; off > 0; off >>= 1) {
    sum += __shfl_xor(sum, off, 32);
    sq  += __shfl_xor(sq,  off, 32);
  }
  __shared__ float red[16];
  const int wave = tid >> 5, lane = tid & 31;
  if (lane == 0) { red[wave] = sum; red[8 + wave] = sq; }
  __syncthreads();
  float ts = 0.0f, tq = 0.0f;
  #pragma unroll
  for (int i = 0; i < 8; ++i) { ts += red[i]; tq += red[8 + i]; }

  const float inv = 1.0f / (float)DIN;
  const float mu  = ts * inv;
  const float var = tq * inv - mu * mu;
  const float rs  = rsqrtf(var + 1e-5f);

  _Float16* arow = A + (size_t)m * DIN;
  #pragma unroll
  for (int q = 0; q < 15; ++q) {
    const int d = q * 256 + tid;                 // d = c*256 + p1*16 + p2
    const float a = (v[q] - mu) * rs * gamma[d] + beta[d];
    arow[d] = (_Float16)a;
  }
}

// ---------------------------------------------------------------------------
// K3: WMMA GEMM. Block = 256 threads (8 waves) computes a 128x128 output tile.
// Wave grid 4(M) x 2(N); each wave owns 32x64 = 2x4 16x16 subtiles.
// K staged through LDS in chunks of 32 halves, row stride padded to 40 halves
// (80 B) so the 16-row fragment gathers are bank-conflict free.
// ---------------------------------------------------------------------------
#define BM  128
#define BN  128
#define KC  32
#define SST 40   // LDS row stride in halves (32 + 8 pad)

__global__ __launch_bounds__(256) void k_gemm_wmma(const _Float16* __restrict__ Ag,  // [M, DIN]
                                                   const _Float16* __restrict__ Bt,  // [DMODEL, DIN]
                                                   const float* __restrict__ bias,   // [DMODEL]
                                                   float* __restrict__ out) {        // [M, DMODEL]
  __shared__ __align__(16) _Float16 As[BM * SST];
  __shared__ __align__(16) _Float16 Bs[BN * SST];

  const int tid  = threadIdx.x;
  const int mblk = blockIdx.x * BM;
  const int nblk = blockIdx.y * BN;

  const int wave = tid >> 5;
  const int lane = tid & 31;
  const int wm = wave >> 1;      // 0..3 : 32-row strip
  const int wn = wave & 1;       // 0..1 : 64-col strip
  const int lr = lane & 15;      // row/col inside a 16x16 tile
  const int g  = lane >> 4;      // K half-group

  v8f acc[2][4];
  {
    v8f z = {};
    #pragma unroll
    for (int i = 0; i < 2; ++i)
      #pragma unroll
      for (int j = 0; j < 4; ++j) acc[i][j] = z;
  }

  // staging assignment: 2 threads per row, 16 halves (32 B) each
  const int lrow  = tid >> 1;          // 0..127
  const int lpart = tid & 1;           // 0 / 1
  const _Float16* aG = Ag + (size_t)(mblk + lrow) * DIN + lpart * 16;
  const _Float16* bG = Bt + (size_t)(nblk + lrow) * DIN + lpart * 16;
  _Float16* aS = &As[lrow * SST + lpart * 16];
  _Float16* bS = &Bs[lrow * SST + lpart * 16];

  for (int k0 = 0; k0 < DIN; k0 += KC) {
    // global -> LDS (16B vector ops)
    *(v8h*)(aS)     = *(const v8h*)(aG + k0);
    *(v8h*)(aS + 8) = *(const v8h*)(aG + k0 + 8);
    *(v8h*)(bS)     = *(const v8h*)(bG + k0);
    *(v8h*)(bS + 8) = *(const v8h*)(bG + k0 + 8);
    if (k0 + KC < DIN) {                       // global_prefetch_b8 next chunk
      __builtin_prefetch(aG + k0 + KC, 0, 3);
      __builtin_prefetch(bG + k0 + KC, 0, 3);
    }
    __syncthreads();

    // A fragment: lane holds row M=lr, K in [8g,8g+8) U [16+8g,16+8g+8)
    v16h af[2];
    #pragma unroll
    for (int i = 0; i < 2; ++i) {
      const _Float16* p = &As[(wm * 32 + i * 16 + lr) * SST];
      v8h lo = *(const v8h*)(p + 8 * g);
      v8h hi = *(const v8h*)(p + 16 + 8 * g);
      af[i] = __builtin_shufflevector(lo, hi, 0, 1, 2, 3, 4, 5, 6, 7,
                                              8, 9, 10, 11, 12, 13, 14, 15);
    }
    // B fragment: lane holds col N=lr, K in [8g,8g+8) U [16+8g,16+8g+8)
    v16h bf[4];
    #pragma unroll
    for (int j = 0; j < 4; ++j) {
      const _Float16* p = &Bs[(wn * 64 + j * 16 + lr) * SST];
      v8h lo = *(const v8h*)(p + 8 * g);
      v8h hi = *(const v8h*)(p + 16 + 8 * g);
      bf[j] = __builtin_shufflevector(lo, hi, 0, 1, 2, 3, 4, 5, 6, 7,
                                              8, 9, 10, 11, 12, 13, 14, 15);
    }

    #pragma unroll
    for (int i = 0; i < 2; ++i)
      #pragma unroll
      for (int j = 0; j < 4; ++j)
        acc[i][j] = __builtin_amdgcn_wmma_f32_16x16x32_f16(
            false, af[i], false, bf[j], (short)0, acc[i][j], false, false);

    __syncthreads();
  }

  // epilogue: C layout -> VGPR e holds M = e + 8*g, N = lr ; fuse bias
  #pragma unroll
  for (int i = 0; i < 2; ++i) {
    const int mrow0 = mblk + wm * 32 + i * 16 + g * 8;
    #pragma unroll
    for (int j = 0; j < 4; ++j) {
      const int n = nblk + wn * 64 + j * 16 + lr;
      const float bb = bias[n];
      float* op = out + (size_t)mrow0 * DMODEL + n;
      #pragma unroll
      for (int e = 0; e < 8; ++e)
        op[(size_t)e * DMODEL] = acc[i][j][e] + bb;
    }
  }
}

// ---------------------------------------------------------------------------
extern "C" void kernel_launch(void* const* d_in, const int* in_sizes, int n_in,
                              void* d_out, int out_size, void* d_ws, size_t ws_size,
                              hipStream_t stream) {
  const float* x     = (const float*)d_in[0];   // [64,3,224,224]
  const float* gamma = (const float*)d_in[1];   // [3840]
  const float* beta  = (const float*)d_in[2];   // [3840]
  const float* w     = (const float*)d_in[3];   // [3840,768]
  const float* bias  = (const float*)d_in[4];   // [768]
  float* out = (float*)d_out;                   // [64,196,768]

  _Float16* A  = (_Float16*)d_ws;                                   // 96.3 MB
  _Float16* Wt = (_Float16*)((char*)d_ws +
                             (size_t)M_TOTAL * DIN * sizeof(_Float16)); // 5.9 MB

  k_wt_convert<<<(DMODEL * DIN) / 256, 256, 0, stream>>>(w, Wt);
  k_patch_ln<<<M_TOTAL, 256, 0, stream>>>(x, gamma, beta, A);
  dim3 grid(M_TOTAL / BM, DMODEL / BN);   // 98 x 6
  k_gemm_wmma<<<grid, 256, 0, stream>>>(A, Wt, bias, out);
}
